// CausalSelfAttention_15822659519025
// MI455X (gfx1250) — matrix operand order repaired and checked by reading, not compiled, not run on hardware
//
#include <hip/hip_runtime.h>
#include <hip/hip_bf16.h>

// Problem constants: B=4, T=2048, D=1024, H=16, HD=64
typedef __bf16 bf16_t;
typedef __attribute__((ext_vector_type(4)))  __bf16 v4bf;
typedef __attribute__((ext_vector_type(8)))  __bf16 v8bf;
typedef __attribute__((ext_vector_type(16))) __bf16 v16bf;
typedef __attribute__((ext_vector_type(8)))  float  v8f;

// ---------- WMMA fragment helpers (layouts per CDNA5 ISA §7.12.2) ----------
__device__ __forceinline__ v16bf frag_a16(const bf16_t* row, int half) {
  v8bf lo = *(const v8bf*)(row + half * 8);
  v8bf hi = *(const v8bf*)(row + 16 + half * 8);
  return __builtin_shufflevector(lo, hi, 0,1,2,3,4,5,6,7,8,9,10,11,12,13,14,15);
}
__device__ __forceinline__ v16bf frag_b16(const bf16_t* col, int half) {
  v8bf lo = *(const v8bf*)(col + half * 8);
  v8bf hi = *(const v8bf*)(col + 16 + half * 8);
  return __builtin_shufflevector(lo, hi, 0,1,2,3,4,5,6,7,8,9,10,11,12,13,14,15);
}
__device__ __forceinline__ v8f wmma_bf16f32(v16bf a, v16bf b, v8f c) {
  return __builtin_amdgcn_wmma_f32_16x16x32_bf16(false, a, false, b, (short)0, c, false, false);
}

// ---------- CDNA5 async global->LDS copy (GLOBAL_LOAD_ASYNC_TO_LDS_B64, ASYNCcnt) ----
__device__ __forceinline__ void async_copy_b64(uint32_t lds_byte_off,
                                               const void* sbase, int voff) {
  asm volatile("global_load_async_to_lds_b64 %0, %1, %2"
               :: "v"(lds_byte_off), "v"(voff), "s"(sbase) : "memory");
}
__device__ __forceinline__ void wait_asynccnt0()  { asm volatile("s_wait_asynccnt 0x0"  ::: "memory"); }
__device__ __forceinline__ void wait_asynccnt16() { asm volatile("s_wait_asynccnt 0x10" ::: "memory"); }

// ---------------------------------------------------------------------------
// Kernel 1: QKV GEMM  X[8192,1024](f32) @ Wqkv[1024,3072](f32) -> Q/K/V bf16 [B,H,T,64]
// 256 threads, 128x128 block tile, K-step 32, register-staged double-buffered LDS.
// ---------------------------------------------------------------------------
struct StageF32 { float4 a[4]; float b[4][4]; };

__device__ __forceinline__ void gload_f32(StageF32& s, const float* __restrict__ x,
                                          const float* __restrict__ w,
                                          int m0, int n0, int kk, int ldw, int tid) {
#pragma unroll
  for (int i = 0; i < 4; ++i) {
    int e = (tid + i * 256) * 4;           // A: 128x32, k fastest, float4
    int kA = e & 31, mA = e >> 5;
    s.a[i] = *(const float4*)&x[(size_t)(m0 + mA) * 1024 + kk + kA];
  }
  int n = tid & 127, kb = (tid >> 7) * 4;  // B: fixed n per thread, 4 k-quads
#pragma unroll
  for (int q = 0; q < 4; ++q)
#pragma unroll
    for (int j = 0; j < 4; ++j)
      s.b[q][j] = w[(size_t)(kk + kb + q * 8 + j) * ldw + n0 + n];
}
__device__ __forceinline__ void sstore_f32(const StageF32& s, bf16_t* A, bf16_t* B, int tid) {
#pragma unroll
  for (int i = 0; i < 4; ++i) {
    int e = (tid + i * 256) * 4;
    int kA = e & 31, mA = e >> 5;
    v4bf t = { (bf16_t)s.a[i].x, (bf16_t)s.a[i].y, (bf16_t)s.a[i].z, (bf16_t)s.a[i].w };
    *(v4bf*)&A[mA * 32 + kA] = t;
  }
  int n = tid & 127, kb = (tid >> 7) * 4;
#pragma unroll
  for (int q = 0; q < 4; ++q) {
    v4bf t = { (bf16_t)s.b[q][0], (bf16_t)s.b[q][1], (bf16_t)s.b[q][2], (bf16_t)s.b[q][3] };
    *(v4bf*)&B[n * 32 + kb + q * 8] = t;
  }
}

__global__ __launch_bounds__(256) void qkv_gemm_kernel(
    const float* __restrict__ x, const float* __restrict__ w,
    bf16_t* __restrict__ qout, bf16_t* __restrict__ kout, bf16_t* __restrict__ vout)
{
  __shared__ __align__(16) bf16_t As[2][128 * 32];
  __shared__ __align__(16) bf16_t Bs[2][128 * 32];

  const int tid = threadIdx.x, lane = tid & 31, wave = tid >> 5;
  const int wm = wave >> 1, wn = wave & 1;
  const int ln = lane & 15, half = lane >> 4;
  const int n0 = blockIdx.x * 128;
  const int m0 = blockIdx.y * 128;

  const int which = n0 >> 10;                 // 0=q,1=k,2=v
  const int nbase = n0 & 1023;
  bf16_t* dst = (which == 0) ? qout : (which == 1 ? kout : vout);

  v8f acc[2][4] = {};
  StageF32 st;

  gload_f32(st, x, w, m0, n0, 0, 3072, tid);
  sstore_f32(st, As[0], Bs[0], tid);
  __syncthreads();

  for (int it = 0; it < 32; ++it) {
    const int cur = it & 1;
    if (it + 1 < 32) gload_f32(st, x, w, m0, n0, (it + 1) * 32, 3072, tid);

    const bf16_t* Ab = As[cur];
    const bf16_t* Bb = Bs[cur];
    v16bf af[2], bfr[4];
#pragma unroll
    for (int sm = 0; sm < 2; ++sm)
      af[sm] = frag_a16(&Ab[(wm * 32 + sm * 16 + ln) * 32], half);
#pragma unroll
    for (int sn = 0; sn < 4; ++sn)
      bfr[sn] = frag_b16(&Bb[(wn * 64 + sn * 16 + ln) * 32], half);
#pragma unroll
    for (int sm = 0; sm < 2; ++sm)
#pragma unroll
      for (int sn = 0; sn < 4; ++sn)
        acc[sm][sn] = wmma_bf16f32(af[sm], bfr[sn], acc[sm][sn]);

    if (it + 1 < 32) sstore_f32(st, As[(it + 1) & 1], Bs[(it + 1) & 1], tid);
    __syncthreads();
  }

#pragma unroll
  for (int sm = 0; sm < 2; ++sm)
#pragma unroll
    for (int sn = 0; sn < 4; ++sn)
#pragma unroll
      for (int r = 0; r < 8; ++r) {
        int mg = m0 + wm * 32 + sm * 16 + r + 8 * half;
        int ng = nbase + wn * 64 + sn * 16 + ln;
        int b = mg >> 11, t = mg & 2047;
        int h = ng >> 6, hd = ng & 63;
        dst[(((size_t)(b * 16 + h)) * 2048 + t) * 64 + hd] = (bf16_t)acc[sm][sn][r];
      }
}

// ---------------------------------------------------------------------------
// Kernel 1.5: V transpose [B,H,T,64] -> [B,H,64,T] (so attention V tiles are
// straight byte copies). Conflict-free LDS pitch 66.
// ---------------------------------------------------------------------------
__global__ __launch_bounds__(128) void vtrans_kernel(const bf16_t* __restrict__ v,
                                                     bf16_t* __restrict__ vt) {
  __shared__ bf16_t tile[64 * 66];
  const int tid = threadIdx.x;
  const int t0 = blockIdx.x * 64;
  const int bh = blockIdx.y;
  const size_t ib = (size_t)bh * 2048 * 64;
  for (int i = tid; i < 4096; i += 128) {
    int hd = i & 63, r = i >> 6;
    tile[r * 66 + hd] = v[ib + (size_t)(t0 + r) * 64 + hd];
  }
  __syncthreads();
  const size_t ob = (size_t)bh * 64 * 2048;
  for (int i = tid; i < 4096; i += 128) {
    int tt = i & 63, hd = i >> 6;
    vt[ob + (size_t)hd * 2048 + t0 + tt] = tile[tt * 66 + hd];
  }
}

// ---------------------------------------------------------------------------
// Kernel 2: flash attention; grid (T/64, B*H), 128 threads (4 waves).
// Q, K, V^T tiles all via GLOBAL_LOAD_ASYNC_TO_LDS_B64; K/V^T double-buffered
// with ASYNCcnt-threshold pipelining (next tile streams in behind compute).
// ---------------------------------------------------------------------------
__global__ __launch_bounds__(128) void attn_kernel(
    const bf16_t* __restrict__ q, const bf16_t* __restrict__ k,
    const bf16_t* __restrict__ vt, bf16_t* __restrict__ o)
{
  __shared__ __align__(16) bf16_t qs[64 * 64];         // [row][hd]
  __shared__ __align__(16) bf16_t ks[2][64 * 64];      // [key][hd]
  __shared__ __align__(16) bf16_t vts[2][64 * 64];     // [hd][key]
  __shared__ __align__(16) bf16_t ps[4][16 * 64];      // per-wave P tile [row][key]

  const int tid = threadIdx.x, lane = tid & 31, wave = tid >> 5;
  const int ln = lane & 15, half = lane >> 4;
  const int qt = blockIdx.x;
  const int bh = blockIdx.y;
  const size_t base = (size_t)bh * 2048 * 64;
  const int q0 = qt * 64;

  const bf16_t* kbase  = k  + base;
  const bf16_t* vtbase = vt + base;                    // [64][2048] for this bh

  // Q tile: 8KB linear async copy
  {
    const uint32_t qdst = (uint32_t)(uintptr_t)(&qs[0]);
    const bf16_t* qsrc = q + base + (size_t)q0 * 64;
#pragma unroll
    for (int i = 0; i < 8; ++i)
      async_copy_b64(qdst + (tid + i * 128) * 8, qsrc, (tid + i * 128) * 8);
  }

  // K tile: contiguous copy; V^T tile: 64 rows of 128B (row stride 4096B).
  auto issue_tile = [&](int j) {
    const int kv0 = j * 64;
    const int buf = j & 1;
    const uint32_t kd = (uint32_t)(uintptr_t)(&ks[buf][0]);
    const uint32_t vd = (uint32_t)(uintptr_t)(&vts[buf][0]);
#pragma unroll
    for (int i = 0; i < 8; ++i) {
      int c = tid + i * 128;                            // 1024 b64 chunks
      async_copy_b64(kd + c * 8, kbase, kv0 * 128 + c * 8);
    }
#pragma unroll
    for (int i = 0; i < 8; ++i) {
      int c = tid + i * 128;
      async_copy_b64(vd + c * 8, vtbase, (c >> 4) * 4096 + (c & 15) * 8 + kv0 * 2);
    }
  };

  issue_tile(0);

  v8f accO[4] = {};
  float mrow[8], lrow[8];
#pragma unroll
  for (int r = 0; r < 8; ++r) { mrow[r] = -1e30f; lrow[r] = 0.f; }

  v16bf aq0 = {}, aq1 = {};

  for (int j = 0; j <= qt; ++j) {                      // lower triangle only
    const int kv0 = j * 64;
    const int buf = j & 1;
    if (j + 1 <= qt) { issue_tile(j + 1); wait_asynccnt16(); }  // tile j done, j+1 in flight
    else             { wait_asynccnt0(); }
    __syncthreads();                                   // tile j visible to all waves
    if (j == 0) {
      aq0 = frag_a16(&qs[(wave * 16 + ln) * 64 + 0], half);
      aq1 = frag_a16(&qs[(wave * 16 + ln) * 64 + 32], half);
    }
    const bf16_t* kb = &ks[buf][0];
    const bf16_t* vb = &vts[buf][0];

    // S = Q K^T  (16 rows x 64 keys per wave)
    v8f s[4] = {};
#pragma unroll
    for (int kt = 0; kt < 4; ++kt) {
      v16bf bk0 = frag_b16(&kb[(kt * 16 + ln) * 64 + 0], half);
      v16bf bk1 = frag_b16(&kb[(kt * 16 + ln) * 64 + 32], half);
      s[kt] = wmma_bf16f32(aq0, bk0, s[kt]);
      s[kt] = wmma_bf16f32(aq1, bk1, s[kt]);
    }

    const bool diag = (j == qt);
#pragma unroll
    for (int kt = 0; kt < 4; ++kt)
#pragma unroll
      for (int r = 0; r < 8; ++r) {
        float sv = s[kt][r] * 0.125f;                  // 1/sqrt(64)
        if (diag) {
          int kg = kv0 + kt * 16 + ln;
          int rg = q0 + wave * 16 + r + 8 * half;
          if (kg > rg) sv = -1e30f;                    // causal mask
        }
        s[kt][r] = sv;
      }

    // online softmax; row reductions across the 16-lane half (wave32)
#pragma unroll
    for (int r = 0; r < 8; ++r) {
      float rmax = s[0][r];
#pragma unroll
      for (int kt = 1; kt < 4; ++kt) rmax = fmaxf(rmax, s[kt][r]);
      rmax = fmaxf(rmax, __shfl_xor(rmax, 1, 32));
      rmax = fmaxf(rmax, __shfl_xor(rmax, 2, 32));
      rmax = fmaxf(rmax, __shfl_xor(rmax, 4, 32));
      rmax = fmaxf(rmax, __shfl_xor(rmax, 8, 32));
      float mnew = fmaxf(mrow[r], rmax);
      float corr = __expf(mrow[r] - mnew);
      float rsum = 0.f;
#pragma unroll
      for (int kt = 0; kt < 4; ++kt) {
        float p = __expf(s[kt][r] - mnew);
        s[kt][r] = p;
        rsum += p;
      }
      rsum += __shfl_xor(rsum, 1, 32);
      rsum += __shfl_xor(rsum, 2, 32);
      rsum += __shfl_xor(rsum, 4, 32);
      rsum += __shfl_xor(rsum, 8, 32);
      lrow[r] = lrow[r] * corr + rsum;
      mrow[r] = mnew;
#pragma unroll
      for (int nt = 0; nt < 4; ++nt) accO[nt][r] *= corr;
    }

    // C-layout P -> A-layout via per-wave LDS bounce
    bf16_t* pw = &ps[wave][0];
#pragma unroll
    for (int kt = 0; kt < 4; ++kt)
#pragma unroll
      for (int r = 0; r < 8; ++r)
        pw[(r + 8 * half) * 64 + kt * 16 + ln] = (bf16_t)s[kt][r];

    v16bf ap0 = frag_a16(&pw[ln * 64 + 0], half);
    v16bf ap1 = frag_a16(&pw[ln * 64 + 32], half);
#pragma unroll
    for (int nt = 0; nt < 4; ++nt) {
      v16bf bv0 = frag_b16(&vb[(nt * 16 + ln) * 64 + 0], half);
      v16bf bv1 = frag_b16(&vb[(nt * 16 + ln) * 64 + 32], half);
      accO[nt] = wmma_bf16f32(ap0, bv0, accO[nt]);
      accO[nt] = wmma_bf16f32(ap1, bv1, accO[nt]);
    }
    __syncthreads();                                   // readers done before buf reuse
  }

  // normalize and write O as bf16 [B,T,D] with D = h*64 + hd
  const int b = bh >> 4, h = bh & 15;
#pragma unroll
  for (int r = 0; r < 8; ++r) {
    float inv = 1.0f / lrow[r];
    int t = q0 + wave * 16 + r + 8 * half;
#pragma unroll
    for (int nt = 0; nt < 4; ++nt)
      o[((size_t)(b * 2048 + t)) * 1024 + h * 64 + nt * 16 + ln] =
          (bf16_t)(accO[nt][r] * inv);
  }
}

// ---------------------------------------------------------------------------
// Kernel 3: projection  O[8192,1024](bf16) @ Wproj[1024,1024](f32) -> out f32
// ---------------------------------------------------------------------------
struct StageP { v4bf a[4]; float b[4][4]; };

__device__ __forceinline__ void gload_p(StageP& s, const bf16_t* __restrict__ a,
                                        const float* __restrict__ w,
                                        int m0, int n0, int kk, int tid) {
#pragma unroll
  for (int i = 0; i < 4; ++i) {
    int e = (tid + i * 256) * 4;
    int kA = e & 31, mA = e >> 5;
    s.a[i] = *(const v4bf*)&a[(size_t)(m0 + mA) * 1024 + kk + kA];
  }
  int n = tid & 127, kb = (tid >> 7) * 4;
#pragma unroll
  for (int q = 0; q < 4; ++q)
#pragma unroll
    for (int j = 0; j < 4; ++j)
      s.b[q][j] = w[(size_t)(kk + kb + q * 8 + j) * 1024 + n0 + n];
}
__device__ __forceinline__ void sstore_p(const StageP& s, bf16_t* A, bf16_t* B, int tid) {
#pragma unroll
  for (int i = 0; i < 4; ++i) {
    int e = (tid + i * 256) * 4;
    int kA = e & 31, mA = e >> 5;
    *(v4bf*)&A[mA * 32 + kA] = s.a[i];
  }
  int n = tid & 127, kb = (tid >> 7) * 4;
#pragma unroll
  for (int q = 0; q < 4; ++q) {
    v4bf t = { (bf16_t)s.b[q][0], (bf16_t)s.b[q][1], (bf16_t)s.b[q][2], (bf16_t)s.b[q][3] };
    *(v4bf*)&B[n * 32 + kb + q * 8] = t;
  }
}

__global__ __launch_bounds__(256) void proj_gemm_kernel(
    const bf16_t* __restrict__ a, const float* __restrict__ w, float* __restrict__ out)
{
  __shared__ __align__(16) bf16_t As[2][128 * 32];
  __shared__ __align__(16) bf16_t Bs[2][128 * 32];

  const int tid = threadIdx.x, lane = tid & 31, wave = tid >> 5;
  const int wm = wave >> 1, wn = wave & 1;
  const int ln = lane & 15, half = lane >> 4;
  const int n0 = blockIdx.x * 128;
  const int m0 = blockIdx.y * 128;

  v8f acc[2][4] = {};
  StageP st;

  gload_p(st, a, w, m0, n0, 0, tid);
  sstore_p(st, As[0], Bs[0], tid);
  __syncthreads();

  for (int it = 0; it < 32; ++it) {
    const int cur = it & 1;
    if (it + 1 < 32) gload_p(st, a, w, m0, n0, (it + 1) * 32, tid);

    const bf16_t* Ab = As[cur];
    const bf16_t* Bb = Bs[cur];
    v16bf af[2], bfr[4];
#pragma unroll
    for (int sm = 0; sm < 2; ++sm)
      af[sm] = frag_a16(&Ab[(wm * 32 + sm * 16 + ln) * 32], half);
#pragma unroll
    for (int sn = 0; sn < 4; ++sn)
      bfr[sn] = frag_b16(&Bb[(wn * 64 + sn * 16 + ln) * 32], half);
#pragma unroll
    for (int sm = 0; sm < 2; ++sm)
#pragma unroll
      for (int sn = 0; sn < 4; ++sn)
        acc[sm][sn] = wmma_bf16f32(af[sm], bfr[sn], acc[sm][sn]);

    if (it + 1 < 32) sstore_p(st, As[(it + 1) & 1], Bs[(it + 1) & 1], tid);
    __syncthreads();
  }

#pragma unroll
  for (int sm = 0; sm < 2; ++sm)
#pragma unroll
    for (int sn = 0; sn < 4; ++sn)
#pragma unroll
      for (int r = 0; r < 8; ++r) {
        int mg = m0 + wm * 32 + sm * 16 + r + 8 * half;
        int ng = n0 + wn * 64 + sn * 16 + ln;
        out[(size_t)mg * 1024 + ng] = acc[sm][sn][r];
      }
}

// ---------------------------------------------------------------------------
extern "C" void kernel_launch(void* const* d_in, const int* in_sizes, int n_in,
                              void* d_out, int out_size, void* d_ws, size_t ws_size,
                              hipStream_t stream) {
  const float* x      = (const float*)d_in[0];   // [4,2048,1024]
  const float* w_qkv  = (const float*)d_in[1];   // [1024,3072]
  const float* w_proj = (const float*)d_in[2];   // [1024,1024]
  float* out = (float*)d_out;                    // [4,2048,1024]

  const size_t per = (size_t)4 * 16 * 2048 * 64; // B*H*T*HD elements
  bf16_t* qb  = (bf16_t*)d_ws;
  bf16_t* kb  = qb + per;
  bf16_t* vb  = kb + per;
  bf16_t* vtb = vb + per;                        // [B,H,64,T] bf16
  bf16_t* ob  = vb;                              // alias: vb is dead after vtrans

  qkv_gemm_kernel <<<dim3(24, 64), 256, 0, stream>>>(x, w_qkv, qb, kb, vb);
  vtrans_kernel   <<<dim3(32, 64), 128, 0, stream>>>(vb, vtb);
  attn_kernel     <<<dim3(32, 64), 128, 0, stream>>>(qb, kb, vtb, ob);
  proj_gemm_kernel<<<dim3(8, 64),  256, 0, stream>>>(ob, w_proj, out);
}
